// SlotClassifiers_274877907651
// MI455X (gfx1250) — matrix operand order repaired and checked by reading, compiled, not run on hardware
//
#include <hip/hip_runtime.h>
#include <hip/hip_bf16.h>

#define BB 8192
#define KK 16
#define DD 1024
#define CC 512
#define LDP (CC + 4)   // padded row stride: 8*516 % 64 == 32 -> no half-wave bank conflict

typedef __attribute__((ext_vector_type(16))) __bf16 v16bf;
typedef __attribute__((ext_vector_type(8)))  __bf16 v8bf;
typedef __attribute__((ext_vector_type(8)))  float  v8f;

__device__ __forceinline__ v16bf cvt16(v8f lo, v8f hi) {
    v16bf r;
#pragma unroll
    for (int i = 0; i < 8; ++i) r[i] = (__bf16)lo[i];
#pragma unroll
    for (int i = 0; i < 8; ++i) r[8 + i] = (__bf16)hi[i];
    return r;
}

__device__ __forceinline__ v16bf cat16(v8bf lo, v8bf hi) {
    return __builtin_shufflevector(lo, hi, 0, 1, 2, 3, 4, 5, 6, 7,
                                           8, 9, 10, 11, 12, 13, 14, 15);
}

// ---------------- bucketing ----------------
__global__ void init_ws(int* counts, int* cursor) {
    int t = threadIdx.x;
    if (t < KK) { counts[t] = 0; cursor[t] = 0; }
}

__global__ void count_slots(const int* __restrict__ slot_ids, int* counts) {
    int b = blockIdx.x * blockDim.x + threadIdx.x;
    if (b < BB) atomicAdd(&counts[slot_ids[b]], 1);
}

__global__ void prefix16(const int* __restrict__ counts, int* offsets) {
    if (threadIdx.x == 0) {
        int acc = 0;
        for (int s = 0; s < KK; ++s) { offsets[s] = acc; acc += counts[s]; }
    }
}

__global__ void scatter_slots(const int* __restrict__ slot_ids,
                              const int* __restrict__ offsets,
                              int* cursor, int* perm) {
    int b = blockIdx.x * blockDim.x + threadIdx.x;
    if (b < BB) {
        int s = slot_ids[b];
        int p = atomicAdd(&cursor[s], 1);
        perm[offsets[s] + p] = b;
    }
}

// ---------------- one-shot bf16 pre-conversion ----------------
__global__ void convert_W(const float* __restrict__ Wp, __bf16* __restrict__ Wbf) {
    size_t i = ((size_t)blockIdx.x * blockDim.x + threadIdx.x) * 8;
    if (i < (size_t)KK * CC * DD) {
        v8f f = *(const v8f*)(Wp + i);
        v8bf o;
#pragma unroll
        for (int j = 0; j < 8; ++j) o[j] = (__bf16)f[j];
        *(v8bf*)(Wbf + i) = o;
    }
}

// Xbf[b,:] = bf16(concepts[b, slot_ids[b], :])  (one block per example row)
__global__ void gather_convert_X(const float* __restrict__ concepts,
                                 const int* __restrict__ slot_ids,
                                 __bf16* __restrict__ Xbf) {
    const int b = blockIdx.x;
    const int s = slot_ids[b];
    const float* src = concepts + ((size_t)b * KK + s) * DD;
    __bf16* dst = Xbf + (size_t)b * DD;
    const int i = threadIdx.x * 8;   // 128 threads * 8 = 1024
    v8f f = *(const v8f*)(src + i);
    v8bf o;
#pragma unroll
    for (int j = 0; j < 8; ++j) o[j] = (__bf16)f[j];
    *(v8bf*)(dst + i) = o;
}

// ---------------- shared epilogue: bias + logsumexp + CE ----------------
__device__ __forceinline__ void epilogue_ce(float* lds, const v8f* acc,
                                            const float* __restrict__ biasp,
                                            const int* __restrict__ labels,
                                            const int* __restrict__ perm,
                                            float* __restrict__ ce,
                                            int s, int base, int valid,
                                            int wv, int lane, int lrow, int half) {
    // C/D layout: VGPR r -> M = r + 8*half, N = lrow.  Add bias on the way to LDS.
#pragma unroll
    for (int t = 0; t < 4; ++t) {
        const int c = (wv * 4 + t) * 16 + lrow;
        const float bv = biasp[s * CC + c];
#pragma unroll
        for (int r = 0; r < 8; ++r)
            lds[(r + 8 * half) * LDP + c] = acc[t][r] + bv;
    }
    __syncthreads();

    // Each wave: logsumexp + CE for rows 2*wv and 2*wv+1 (fixed-order -> deterministic)
#pragma unroll
    for (int rr = 0; rr < 2; ++rr) {
        const int m = 2 * wv + rr;
        if (m >= valid) continue;
        const float* row = lds + m * LDP;
        float mx = -3.402823466e38f;
        for (int i = lane; i < CC; i += 32) mx = fmaxf(mx, row[i]);
#pragma unroll
        for (int o = 16; o > 0; o >>= 1) mx = fmaxf(mx, __shfl_xor(mx, o, 32));
        float se = 0.0f;
        for (int i = lane; i < CC; i += 32) se += expf(row[i] - mx);
#pragma unroll
        for (int o = 16; o > 0; o >>= 1) se += __shfl_xor(se, o, 32);
        if (lane == 0) {
            const int b2  = perm[base + m];
            const int lab = labels[b2];
            ce[b2] = (mx + logf(se)) - row[lab];   // -log p[label]
        }
    }
}

// ---------------- fast path: pure-bf16 WMMA GEMM + CE ----------------
__global__ __launch_bounds__(256)
void slot_gemm_ce_bf16(const __bf16* __restrict__ Xbf,
                       const __bf16* __restrict__ Wbf,
                       const float*  __restrict__ biasp,
                       const int*    __restrict__ labels,
                       const int*    __restrict__ perm,
                       const int*    __restrict__ counts,
                       const int*    __restrict__ offsets,
                       float*        __restrict__ ce) {
    const int s     = blockIdx.y;
    const int mtile = blockIdx.x;
    const int cnt   = counts[s];
    if (mtile * 16 >= cnt) return;                 // uniform exit, EXEC all-1s for WMMA
    const int base  = offsets[s] + mtile * 16;
    const int valid = min(16, cnt - mtile * 16);

    __shared__ float lds[16 * LDP];

    const int tid  = threadIdx.x;
    const int lane = tid & 31;
    const int wv   = tid >> 5;
    const int lrow = lane & 15;
    const int half = lane >> 4;

    const int arow = (lrow < valid) ? lrow : (valid - 1);
    const int bex  = perm[base + arow];
    const __bf16* aptr  = Xbf + (size_t)bex * DD;
    const __bf16* wbase = Wbf + (size_t)s * CC * DD;

    v8f acc[4];
#pragma unroll
    for (int t = 0; t < 4; ++t)
#pragma unroll
        for (int r = 0; r < 8; ++r) acc[t][r] = 0.0f;

#pragma unroll 2
    for (int kk = 0; kk < DD / 32; ++kk) {
        // A 16x32 bf16 layout: lane holds K = 8*half+0..7 and +16..23 of row lrow
        const int d0 = kk * 32 + 8 * half;
        v16bf af = cat16(*(const v8bf*)(aptr + d0),
                         *(const v8bf*)(aptr + d0 + 16));
        // B 32x16 bf16 layout: lane holds K = 8*half+0..7 and +16..23 of column lrow
        const int db = kk * 32 + 8 * half;
#pragma unroll
        for (int t = 0; t < 4; ++t) {
            const int c = (wv * 4 + t) * 16 + lrow;
            const __bf16* bp = wbase + (size_t)c * DD + db;
            v16bf bf = cat16(*(const v8bf*)(bp), *(const v8bf*)(bp + 16));
            acc[t] = __builtin_amdgcn_wmma_f32_16x16x32_bf16(
                false, af, false, bf, (short)0, acc[t], false, false);
        }
    }

    epilogue_ce(lds, acc, biasp, labels, perm, ce, s, base, valid, wv, lane, lrow, half);
}

// ---------------- fallback path: inline-convert WMMA GEMM + CE ----------------
__global__ __launch_bounds__(256)
void slot_gemm_ce_f32(const float* __restrict__ concepts,
                      const float* __restrict__ Wp,
                      const float* __restrict__ biasp,
                      const int*   __restrict__ labels,
                      const int*   __restrict__ perm,
                      const int*   __restrict__ counts,
                      const int*   __restrict__ offsets,
                      float*       __restrict__ ce) {
    const int s     = blockIdx.y;
    const int mtile = blockIdx.x;
    const int cnt   = counts[s];
    if (mtile * 16 >= cnt) return;
    const int base  = offsets[s] + mtile * 16;
    const int valid = min(16, cnt - mtile * 16);

    __shared__ float lds[16 * LDP];

    const int tid  = threadIdx.x;
    const int lane = tid & 31;
    const int wv   = tid >> 5;
    const int lrow = lane & 15;
    const int half = lane >> 4;

    const int arow = (lrow < valid) ? lrow : (valid - 1);
    const int bex  = perm[base + arow];
    const float* aptr  = concepts + ((size_t)bex * KK + s) * DD;
    const float* wbase = Wp + (size_t)s * CC * DD;

    v8f acc[4];
#pragma unroll
    for (int t = 0; t < 4; ++t)
#pragma unroll
        for (int r = 0; r < 8; ++r) acc[t][r] = 0.0f;

#pragma unroll 2
    for (int kk = 0; kk < DD / 32; ++kk) {
        const int d0 = kk * 32 + 8 * half;
        v16bf af = cvt16(*(const v8f*)(aptr + d0),
                         *(const v8f*)(aptr + d0 + 16));
        const int db = kk * 32 + 8 * half;
#pragma unroll
        for (int t = 0; t < 4; ++t) {
            const int c = (wv * 4 + t) * 16 + lrow;
            const float* bp = wbase + (size_t)c * DD + db;
            v16bf bf = cvt16(*(const v8f*)(bp), *(const v8f*)(bp + 16));
            acc[t] = __builtin_amdgcn_wmma_f32_16x16x32_bf16(
                false, af, false, bf, (short)0, acc[t], false, false);
        }
    }

    epilogue_ce(lds, acc, biasp, labels, perm, ce, s, base, valid, wv, lane, lrow, half);
}

// ---------------- deterministic per-slot mean + final scalar ----------------
__global__ void slot_mean(const float* __restrict__ ce,
                          const int* __restrict__ slot_ids,
                          const int* __restrict__ counts,
                          float* means) {
    const int s = blockIdx.x;
    __shared__ float red[256];
    float acc = 0.0f;
    for (int b = threadIdx.x; b < BB; b += 256)
        if (slot_ids[b] == s) acc += ce[b];
    red[threadIdx.x] = acc;
    __syncthreads();
    for (int st = 128; st > 0; st >>= 1) {
        if (threadIdx.x < st) red[threadIdx.x] += red[threadIdx.x + st];
        __syncthreads();
    }
    if (threadIdx.x == 0) {
        const int c = counts[s];
        means[s] = red[0] / fmaxf((float)c, 1.0f);
    }
}

__global__ void finalize(const float* __restrict__ means,
                         const int* __restrict__ counts,
                         float* out) {
    if (threadIdx.x == 0) {
        float sum = 0.0f; int np = 0;
        for (int s = 0; s < KK; ++s)
            if (counts[s] > 0) { sum += means[s]; ++np; }
        out[0] = sum / (float)max(np, 1);
    }
}

extern "C" void kernel_launch(void* const* d_in, const int* in_sizes, int n_in,
                              void* d_out, int out_size, void* d_ws, size_t ws_size,
                              hipStream_t stream) {
    const float* concepts = (const float*)d_in[0];
    const int*   slot_ids = (const int*)d_in[1];
    const int*   labels   = (const int*)d_in[2];
    const float* W        = (const float*)d_in[3];
    const float* bias     = (const float*)d_in[4];
    float* out = (float*)d_out;

    // scratch layout
    int*   counts  = (int*)d_ws;            // 16
    int*   offsets = counts + 16;           // 16
    int*   cursor  = offsets + 16;          // 16
    int*   perm    = cursor + 16;           // 8192
    float* ce      = (float*)(perm + BB);   // 8192
    float* means   = ce + BB;               // 16
    const size_t fixed_bytes = ((48 + BB) * sizeof(int) + (BB + 16) * sizeof(float) + 255) & ~(size_t)255;
    __bf16* Xbf = (__bf16*)((char*)d_ws + fixed_bytes);          // BB*DD bf16   (16 MB)
    __bf16* Wbf = Xbf + (size_t)BB * DD;                         // KK*CC*DD bf16 (16 MB)
    const size_t need = fixed_bytes + ((size_t)BB * DD + (size_t)KK * CC * DD) * sizeof(__bf16);

    init_ws<<<1, 64, 0, stream>>>(counts, cursor);
    count_slots<<<BB / 256, 256, 0, stream>>>(slot_ids, counts);
    prefix16<<<1, 32, 0, stream>>>(counts, offsets);
    scatter_slots<<<BB / 256, 256, 0, stream>>>(slot_ids, offsets, cursor, perm);

    dim3 grid(BB / 16, KK);
    if (ws_size >= need) {
        // fast path: one-shot bf16 conversion, pure load+WMMA inner loop
        const size_t wElems = (size_t)KK * CC * DD;              // 8M
        convert_W<<<(int)(wElems / 8 / 256), 256, 0, stream>>>(W, Wbf);
        gather_convert_X<<<BB, 128, 0, stream>>>(concepts, slot_ids, Xbf);
        slot_gemm_ce_bf16<<<grid, 256, 0, stream>>>(Xbf, Wbf, bias, labels,
                                                    perm, counts, offsets, ce);
    } else {
        slot_gemm_ce_f32<<<grid, 256, 0, stream>>>(concepts, W, bias, labels,
                                                   perm, counts, offsets, ce);
    }

    slot_mean<<<KK, 256, 0, stream>>>(ce, slot_ids, counts, means);
    finalize<<<1, 32, 0, stream>>>(means, counts, out);
}
